// MinGRUBlock_67259187855489
// MI455X (gfx1250) — compile-verified, matrix operand order checked by reading, not run on hardware
//
#include <hip/hip_runtime.h>
#include <hip/hip_bf16.h>
#include <math.h>

// Problem constants (from reference): B=8, T=4096, D=1024, H=1024
#define B_DIM 8
#define T_DIM 4096
#define D_DIM 1024
#define H_DIM 1024
#define LN_EPS 1e-5f

// scan chunking: 64 chunks of 64 timesteps
#define N_CH 64
#define L_CH 64

typedef __attribute__((ext_vector_type(16))) __bf16 v16bf;
typedef __attribute__((ext_vector_type(8)))  float  v8f;

// float -> bf16 bits, round-to-nearest-even
__device__ __forceinline__ unsigned short f2bf(float f) {
  union { float f; unsigned u; } v; v.f = f;
  unsigned r = v.u + 0x7FFFu + ((v.u >> 16) & 1u);
  return (unsigned short)(r >> 16);
}

// pack two floats into one dword of bf16 (lo, hi)
__device__ __forceinline__ unsigned pack_bf16(float lo, float hi) {
#if __has_builtin(__builtin_amdgcn_cvt_pk_bf16_f32)
  auto t = __builtin_amdgcn_cvt_pk_bf16_f32(lo, hi);   // v_cvt_pk_bf16_f32
  unsigned u; __builtin_memcpy(&u, &t, 4);
  return u;
#else
  return (unsigned)f2bf(lo) | ((unsigned)f2bf(hi) << 16);
#endif
}

__device__ __forceinline__ float softplus_f(float v) {
  return fmaxf(v, 0.0f) + log1pf(__expf(-fabsf(v)));   // stable
}

__device__ __forceinline__ float log_g_f(float a) {
  return (a >= 0.0f) ? __logf(a + 0.5f) : -softplus_f(-a);
}

__device__ __forceinline__ float logaddexp_f(float a, float b) {
  float mx = fmaxf(a, b), mn = fminf(a, b);
  return (mx == -INFINITY) ? -INFINITY : mx + log1pf(__expf(mn - mx));
}

// low 32 bits of a generic pointer to __shared__ = LDS byte address
__device__ __forceinline__ unsigned lds_addr32(const void* p) {
  return (unsigned)(uintptr_t)p;
}

__device__ __forceinline__ void async_copy_b128(unsigned lds, unsigned long long gaddr) {
  asm volatile("global_load_async_to_lds_b128 %0, %1, off"
               :: "v"(lds), "v"(gaddr) : "memory");
}

// ---------------------------------------------------------------------------
// Kernel 0a: x fp32 -> bf16 (row-major unchanged). 8 elements/thread.
// ---------------------------------------------------------------------------
__global__ __launch_bounds__(256)
void cvt_x(const float* __restrict__ x, unsigned short* __restrict__ xb)
{
  size_t i = ((size_t)blockIdx.x * 256 + threadIdx.x) * 8;
  float4 a = *(const float4*)&x[i];
  float4 b = *(const float4*)&x[i + 4];
  uint4 o;
  o.x = pack_bf16(a.x, a.y);
  o.y = pack_bf16(a.z, a.w);
  o.z = pack_bf16(b.x, b.y);
  o.w = pack_bf16(b.z, b.w);
  *(uint4*)&xb[i] = o;
}

// ---------------------------------------------------------------------------
// Kernel 0b: W fp32 [k][n] -> bf16 [n][k] (transpose + convert), via a 64x64
// LDS tile so both global sides stay coalesced. blockIdx.z selects Wz/Wh.
// ---------------------------------------------------------------------------
__global__ __launch_bounds__(256)
void cvt_w(const float* __restrict__ Wz, const float* __restrict__ Wh,
           unsigned short* __restrict__ Wzb, unsigned short* __restrict__ Whb)
{
  __shared__ unsigned short Tb[64 * 64];          // [n][k] bf16
  const float*    src = blockIdx.z ? Wh : Wz;
  unsigned short* dst = blockIdx.z ? Whb : Wzb;
  const int k0  = blockIdx.y * 64;
  const int n1  = blockIdx.x * 64;
  const int tid = threadIdx.x;
  unsigned* TbW = (unsigned*)Tb;

  #pragma unroll
  for (int i = 0; i < 2; ++i) {
    int q  = tid + i * 256;                       // 0..511
    int ng = q & 15;                              // n quad (coalesced)
    int kp = q >> 4;                              // k pair 0..31
    size_t g0 = (size_t)(k0 + 2 * kp) * H_DIM + n1 + 4 * ng;
    float4 r0 = *(const float4*)&src[g0];
    float4 r1 = *(const float4*)&src[g0 + H_DIM];
    int nb = 4 * ng;
    TbW[(nb + 0) * 32 + kp] = pack_bf16(r0.x, r1.x);
    TbW[(nb + 1) * 32 + kp] = pack_bf16(r0.y, r1.y);
    TbW[(nb + 2) * 32 + kp] = pack_bf16(r0.z, r1.z);
    TbW[(nb + 3) * 32 + kp] = pack_bf16(r0.w, r1.w);
  }
  __syncthreads();
  #pragma unroll
  for (int i = 0; i < 2; ++i) {
    int q = tid + i * 256;                        // 0..511
    int n = q >> 3, c = q & 7;
    *(uint4*)&dst[(size_t)(n1 + n) * D_DIM + k0 + 8 * c] =
        *(const uint4*)&Tb[n * 64 + 8 * c];
  }
}

// ---------------------------------------------------------------------------
// Kernel 1: fused dual-GEMM + activation epilogue, all-bf16 operands.
//   k     = x @ Wz + bz  -> log_coeffs = -softplus(k), log_z = -softplus(-k)
//   g_pre = x @ Wh + bh  -> log_values = log_z + log_g(g_pre)
// Block: 128 threads (4 waves). Output tile 64(M) x 64(N).
// All three tiles arrive via ASYNC DMA (global_load_async_to_lds_b128) into
// double-buffered LDS; next tile's copies overlap with current WMMAs.
// ---------------------------------------------------------------------------
__global__ __launch_bounds__(128)
void mingru_gemm_act(const unsigned short* __restrict__ xb,
                     const unsigned short* __restrict__ Wzb,
                     const unsigned short* __restrict__ Whb,
                     const float* __restrict__ bz, const float* __restrict__ bh,
                     float* __restrict__ log_coeffs,
                     float* __restrict__ log_values)
{
  __shared__ unsigned short As [2][64 * 32];  // A  tile [m][k], bf16
  __shared__ unsigned short Bzs[2][64 * 32];  // Wz tile [n][k], bf16
  __shared__ unsigned short Bhs[2][64 * 32];  // Wh tile [n][k], bf16

  const int tid   = threadIdx.x;
  const int w     = tid >> 5;        // wave id 0..3
  const int lane  = tid & 31;
  const int lrow  = lane & 15;       // M row / N col within fragment
  const int khalf = lane >> 4;       // K-group select (lanes 16-31 -> +8)

  const int m0 = blockIdx.y * 64;    // global row base (B*T dimension)
  const int n0 = blockIdx.x * 64;    // global col base (H dimension)

  const v8f vzero = {0.f,0.f,0.f,0.f,0.f,0.f,0.f,0.f};
  v8f accZ[4] = {vzero, vzero, vzero, vzero};
  v8f accH[4] = {vzero, vzero, vzero, vzero};

  union Frag { v16bf v; unsigned u[8]; };

  // Issue async copies of one 64x32 bf16 tile triple at K offset `ko`.
  // 256 chunks of 16B per tile; each thread issues 2 per tile.
  auto issue_tiles = [&](int ko, int buf) {
    #pragma unroll
    for (int i = 0; i < 2; ++i) {
      int q  = tid + i * 128;                   // 0..255
      int rr = q >> 2;                          // row (m or n) 0..63
      int kc = q & 3;                           // 16B chunk (8 bf16) in K
      unsigned loff = (unsigned)(rr * 32 + 8 * kc);
      async_copy_b128(lds_addr32(&As[buf][loff]),
          (unsigned long long)(uintptr_t)(xb + (size_t)(m0 + rr) * D_DIM + ko + 8 * kc));
      async_copy_b128(lds_addr32(&Bzs[buf][loff]),
          (unsigned long long)(uintptr_t)(Wzb + (size_t)(n0 + rr) * D_DIM + ko + 8 * kc));
      async_copy_b128(lds_addr32(&Bhs[buf][loff]),
          (unsigned long long)(uintptr_t)(Whb + (size_t)(n0 + rr) * D_DIM + ko + 8 * kc));
    }
  };

  issue_tiles(0, 0);

  for (int ko = 0; ko < D_DIM; ko += 32) {
    const int buf = (ko >> 5) & 1;
    // current buffer's DMA must land; align all waves
    asm volatile("s_wait_asynccnt 0x0" ::: "memory");
    __syncthreads();
    if (ko + 32 < D_DIM) {
      issue_tiles(ko + 32, buf ^ 1);            // overlap DMA with compute
      __builtin_prefetch(&xb[(size_t)(m0 + (tid >> 1)) * D_DIM + ko + 64], 0, 1);
    }

    const unsigned* Au  = (const unsigned*)As[buf];
    const unsigned* Bzu = (const unsigned*)Bzs[buf];
    const unsigned* Bhu = (const unsigned*)Bhs[buf];

    // B fragments (shared across the 4 M sub-tiles).
    // 16-bit operand striping: lane holds row/col (lrow), K pairs per dword,
    // k(v,p) = (v%4)*2 + (v/4)*16 + 8*khalf + p -> dword-contiguous runs of 4
    Frag bzf, bhf;
    #pragma unroll
    for (int v = 0; v < 8; ++v) {
      int bk = (v & 3) * 2 + (v >> 2) * 16 + khalf * 8;
      int ci = ((w * 16 + lrow) * 32 + bk) >> 1;   // dword index in LDS tile
      bzf.u[v] = Bzu[ci];
      bhf.u[v] = Bhu[ci];
    }

    #pragma unroll
    for (int mi = 0; mi < 4; ++mi) {
      Frag af;
      #pragma unroll
      for (int v = 0; v < 8; ++v) {
        int bk = (v & 3) * 2 + (v >> 2) * 16 + khalf * 8;
        af.u[v] = Au[((mi * 16 + lrow) * 32 + bk) >> 1];
      }
      accZ[mi] = __builtin_amdgcn_wmma_f32_16x16x32_bf16(
          false, af.v, false, bzf.v, (short)0, accZ[mi], false, false);
      accH[mi] = __builtin_amdgcn_wmma_f32_16x16x32_bf16(
          false, af.v, false, bhf.v, (short)0, accH[mi], false, false);
    }
  }

  // ---- epilogue: bias + log-space activations, write scan inputs ----
  // C/D layout: VGPR j -> M = j (lanes 0-15) / j+8 (lanes 16-31), N = lane&15
  const int n  = n0 + w * 16 + lrow;
  const float bzn = bz[n];
  const float bhn = bh[n];
  #pragma unroll
  for (int mi = 0; mi < 4; ++mi) {
    #pragma unroll
    for (int j = 0; j < 8; ++j) {
      int r = m0 + mi * 16 + khalf * 8 + j;
      float kz = accZ[mi][j] + bzn;
      float ph = accH[mi][j] + bhn;
      float lc = -softplus_f(kz);          // log_coeffs
      float lz = -softplus_f(-kz);         // log_z
      float lv = lz + log_g_f(ph);         // log_values
      size_t off = (size_t)r * H_DIM + n;
      log_coeffs[off] = lc;
      log_values[off] = lv;
    }
  }
}

// ---------------------------------------------------------------------------
// Chunked log-space scan. The recurrence h_t = c_t*h_{t-1} + u_t composes
// associatively: a segment is (la = sum log c, lv = log(result from h_in=0)).
//   combine(prefix, seg): la = la_p + la_s ; lv = logaddexp(lv_p + la_s, lv_s)
// Pass 1: per (b,h,chunk) compute segment state.          524288 threads
// Pass 2: per (b,h) exclusive-scan the 64 chunk states.     8192 threads
// Pass 3: per (b,h,chunk) replay with incoming log_h (=prefix lv), write h.
// ---------------------------------------------------------------------------
__global__ __launch_bounds__(256)
void mingru_scan_p1(const float* __restrict__ lc, const float* __restrict__ lv,
                    float* __restrict__ stA, float* __restrict__ stV)
{
  int gid = blockIdx.x * 256 + threadIdx.x;     // B*H*N_CH
  int h = gid & (H_DIM - 1);
  int c = (gid >> 10) & (N_CH - 1);
  int b = gid >> 16;
  size_t base = ((size_t)b * T_DIM + (size_t)c * L_CH) * H_DIM + h;

  float la = 0.0f, lvs = -INFINITY;
  for (int t = 0; t < L_CH; ++t) {
    size_t off = base + (size_t)t * H_DIM;
    float ct = lc[off];
    float ut = lv[off];
    la += ct;
    lvs = logaddexp_f(lvs + ct, ut);
  }
  size_t sidx = ((size_t)b * N_CH + c) * H_DIM + h;
  stA[sidx] = la;
  stV[sidx] = lvs;
}

__global__ __launch_bounds__(256)
void mingru_scan_p2(float* __restrict__ stA, float* __restrict__ stV)
{
  int gid = blockIdx.x * 256 + threadIdx.x;     // B*H
  int b = gid >> 10;
  int h = gid & (H_DIM - 1);

  float pla = 0.0f, plv = -INFINITY;            // identity prefix
  for (int c = 0; c < N_CH; ++c) {
    size_t idx = ((size_t)b * N_CH + c) * H_DIM + h;
    float la = stA[idx];
    float lvs = stV[idx];
    stA[idx] = pla;                             // exclusive prefix (in-place)
    stV[idx] = plv;
    plv = logaddexp_f(plv + la, lvs);
    pla += la;
  }
}

__global__ __launch_bounds__(256)
void mingru_scan_p3(const float* __restrict__ lc, const float* __restrict__ lv,
                    const float* __restrict__ stV,
                    float* __restrict__ hbuf, float* __restrict__ hidden)
{
  int gid = blockIdx.x * 256 + threadIdx.x;     // B*H*N_CH
  int h = gid & (H_DIM - 1);
  int c = (gid >> 10) & (N_CH - 1);
  int b = gid >> 16;
  size_t base = ((size_t)b * T_DIM + (size_t)c * L_CH) * H_DIM + h;

  // prefix lv IS log(h) entering this chunk (identity -> -inf)
  float logh = stV[((size_t)b * N_CH + c) * H_DIM + h];
  for (int t = 0; t < L_CH; ++t) {
    size_t off = base + (size_t)t * H_DIM;
    logh = logaddexp_f(lc[off] + logh, lv[off]);
    hbuf[off] = __expf(logh);
  }
  if (c == N_CH - 1) {
    hidden[(size_t)b * H_DIM + h] = __expf(logh);
  }
}

// ---------------------------------------------------------------------------
// LayerNorm over H + affine + residual, in-place on the h values in d_out.
// ---------------------------------------------------------------------------
__global__ __launch_bounds__(256)
void mingru_ln(const float* __restrict__ x, const float* __restrict__ gamma,
               const float* __restrict__ beta, float* __restrict__ out)
{
  __shared__ float s1[256];
  __shared__ float s2[256];
  const int r   = blockIdx.x;
  const int tid = threadIdx.x;

  float hv[4];
  float sum = 0.f, sq = 0.f;
  #pragma unroll
  for (int i = 0; i < 4; ++i) {
    int nn = tid + i * 256;
    float v = out[(size_t)r * H_DIM + nn];
    hv[i] = v; sum += v; sq += v * v;
  }
  s1[tid] = sum; s2[tid] = sq;
  __syncthreads();
  for (int sfold = 128; sfold > 0; sfold >>= 1) {
    if (tid < sfold) { s1[tid] += s1[tid + sfold]; s2[tid] += s2[tid + sfold]; }
    __syncthreads();
  }
  float mean = s1[0] * (1.0f / H_DIM);
  float var  = s2[0] * (1.0f / H_DIM) - mean * mean;
  float inv  = rsqrtf(var + LN_EPS);

  #pragma unroll
  for (int i = 0; i < 4; ++i) {
    int nn = tid + i * 256;
    float o = (hv[i] - mean) * inv * gamma[nn] + beta[nn]
            + x[(size_t)r * D_DIM + nn];
    out[(size_t)r * H_DIM + nn] = o;
  }
}

// ---------------------------------------------------------------------------
extern "C" void kernel_launch(void* const* d_in, const int* in_sizes, int n_in,
                              void* d_out, int out_size, void* d_ws, size_t ws_size,
                              hipStream_t stream)
{
  (void)in_sizes; (void)n_in; (void)out_size; (void)ws_size;

  const float* x     = (const float*)d_in[0];
  const float* Wz    = (const float*)d_in[1];
  const float* bz    = (const float*)d_in[2];
  const float* Wh    = (const float*)d_in[3];
  const float* bh    = (const float*)d_in[4];
  const float* gamma = (const float*)d_in[5];
  const float* beta  = (const float*)d_in[6];

  float* out    = (float*)d_out;                               // B*T*H
  float* hidden = out + (size_t)B_DIM * T_DIM * H_DIM;         // B*H tail

  const size_t BTH = (size_t)B_DIM * T_DIM * H_DIM;            // 33.5M
  float* lc  = (float*)d_ws;                                   // log_coeffs
  float* lv  = lc + BTH;                                       // log_values
  float* stA = lv + BTH;                                       // chunk sum(log c)
  float* stV = stA + (size_t)B_DIM * N_CH * H_DIM;             // chunk log value
  unsigned short* xb  = (unsigned short*)(stV + (size_t)B_DIM * N_CH * H_DIM);
  unsigned short* Wzb = xb + (size_t)B_DIM * T_DIM * D_DIM;    // bf16 x
  unsigned short* Whb = Wzb + (size_t)D_DIM * H_DIM;           // bf16 W [n][k]

  // precision prep: bf16 copies (x as-is, weights transposed to [n][k])
  cvt_x<<<(int)(BTH / (256 * 8)), 256, 0, stream>>>(x, xb);
  cvt_w<<<dim3(H_DIM / 64, D_DIM / 64, 2), 256, 0, stream>>>(Wz, Wh, Wzb, Whb);

  dim3 gGemm(H_DIM / 64, (B_DIM * T_DIM) / 64);                // 16 x 512
  mingru_gemm_act<<<gGemm, 128, 0, stream>>>(xb, Wzb, Whb, bz, bh, lc, lv);

  int scanThreads = B_DIM * H_DIM * N_CH;                      // 524288
  mingru_scan_p1<<<scanThreads / 256, 256, 0, stream>>>(lc, lv, stA, stV);
  mingru_scan_p2<<<(B_DIM * H_DIM) / 256, 256, 0, stream>>>(stA, stV);
  mingru_scan_p3<<<scanThreads / 256, 256, 0, stream>>>(lc, lv, stV, out, hidden);

  mingru_ln<<<B_DIM * T_DIM, 256, 0, stream>>>(x, gamma, beta, out);
}
